// AtomAttentionEncoder_5068061409556
// MI455X (gfx1250) — matrix operand order repaired and checked by reading, not compiled, not run on hardware
//
#include <hip/hip_runtime.h>
#include <hip/hip_bf16.h>
#include <cstdint>
#include <cstddef>

// ---------------- problem constants ----------------
#define B_     2
#define N_     8192
#define C_     128
#define CP_    16
#define H_     4
#define NQ_    32
#define NK_    128
#define CT_    384
#define T_     512
#define NBLK_  3
#define NB_    256          // N_/NQ_
#define MT_    16384        // B_*N_
#define DH_    32           // C_/H_
#define FEAT_  389
#define FEATP_ 416          // 13*32

typedef _Float16 v16h __attribute__((ext_vector_type(16)));
typedef _Float16 v8h  __attribute__((ext_vector_type(8)));
typedef float    v8f  __attribute__((ext_vector_type(8)));
typedef float    v2f  __attribute__((ext_vector_type(2)));

union V16 { v16h v; v8h h[2]; };

#if defined(__has_builtin)
#if __has_builtin(__builtin_amdgcn_wmma_f32_16x16x4_f32)
#define HAVE_WMMA_F32X4 1
#endif
#endif

__device__ __forceinline__ float sigmf(float x) { return 1.f / (1.f + __expf(-x)); }

__device__ __forceinline__ v8h v8h_zero() {
    v8h z;
#pragma unroll
    for (int e = 0; e < 8; ++e) z[e] = (_Float16)0.f;
    return z;
}

// ======================================================================
// Generic WMMA GEMM: out[M,N] = A[M,K](f16,row,lda=K) @ W[Kpad,ldw](f16,row)
// W is column-padded to ldw (multiple of 64, zero-filled) so vector loads
// need no guards. Block tile 64x64, 4 waves (16 rows each), K step 32.
// A tile row-major in LDS; W tile stored column-major so both fragments
// read as 2x ds_load_b128.
// ======================================================================
__global__ void k_gemm(const _Float16* __restrict__ A, const _Float16* __restrict__ W,
                       const float* __restrict__ bias, float* __restrict__ outF,
                       _Float16* __restrict__ outH, int M, int K, int N, int ldw,
                       int relu) {
    __shared__ __align__(16) _Float16 As[64 * 40];
    __shared__ __align__(16) _Float16 WsT[64 * 40];
    const int AST = 40, WST = 40;
    int tid = threadIdx.x, lane = tid & 31, wid = tid >> 5;
    int hi = lane >> 4, li = lane & 15;
    int m0 = blockIdx.x * 64, n0 = blockIdx.y * 64;

    // cooperative-load coordinates
    int ar = tid >> 1, ac = (tid & 1) * 16;   // A: row 0..63, col 0/16
    int wr = tid >> 2, wc = (tid & 3) * 16;   // W: k-row 0..31, col seg 0..48

    v8f acc[4];
#pragma unroll
    for (int nt = 0; nt < 4; ++nt)
#pragma unroll
        for (int r = 0; r < 8; ++r) acc[nt][r] = 0.f;

    for (int k0 = 0; k0 < K; k0 += 32) {
        // ---- A tile (vector global load -> vector LDS store) ----
        {
            v8h a0 = v8h_zero(), a1 = a0;
            int m = m0 + ar;
            if (m < M) {
                const _Float16* src = A + (size_t)m * K + k0 + ac;
                a0 = *(const v8h*)src;
                a1 = *(const v8h*)(src + 8);
                if (k0 + 32 < K) __builtin_prefetch(src + 32, 0, 3);
            }
            *(v8h*)&As[ar * AST + ac] = a0;
            *(v8h*)&As[ar * AST + ac + 8] = a1;
        }
        // ---- W tile (vector global load -> transposed LDS scatter) ----
        {
            const _Float16* src = W + (size_t)(k0 + wr) * ldw + n0 + wc;
            v8h w0 = *(const v8h*)src;
            v8h w1 = *(const v8h*)(src + 8);
            if (k0 + 32 < K) __builtin_prefetch(src + 32 * ldw, 0, 3);
#pragma unroll
            for (int e = 0; e < 8; ++e) WsT[(wc + e) * WST + wr] = w0[e];
#pragma unroll
            for (int e = 0; e < 8; ++e) WsT[(wc + 8 + e) * WST + wr] = w1[e];
        }
        __syncthreads();

        int row = wid * 16 + li;
        V16 af;
        af.h[0] = *(const v8h*)&As[row * AST + hi * 8];
        af.h[1] = *(const v8h*)&As[row * AST + 16 + hi * 8];
#pragma unroll
        for (int nt = 0; nt < 4; ++nt) {
            int col = nt * 16 + li;
            V16 bf;
            bf.h[0] = *(const v8h*)&WsT[col * WST + hi * 8];
            bf.h[1] = *(const v8h*)&WsT[col * WST + 16 + hi * 8];
            acc[nt] = __builtin_amdgcn_wmma_f32_16x16x32_f16(false, af.v, false, bf.v,
                                                             (short)0, acc[nt], false, false);
        }
        __syncthreads();
    }
    for (int nt = 0; nt < 4; ++nt) {
        int n = n0 + nt * 16 + li;
        if (n >= N) continue;
        float bv = bias ? bias[n] : 0.f;
#pragma unroll
        for (int r = 0; r < 8; ++r) {
            int m = m0 + wid * 16 + r + 8 * hi;
            if (m >= M) continue;
            float x = acc[nt][r] + bv;
            if (relu) x = fmaxf(x, 0.f);
            size_t o = (size_t)m * N + n;
            if (outF) outF[o] = x;
            if (outH) outH[o] = (_Float16)x;
        }
    }
}

// ======================================================================
// small elementwise / conversion kernels
// ======================================================================
// pad rows to Kpad and cols to Npad with zeros; dst is [Kpad, Npad]
__global__ void k_cvt_w(const float* __restrict__ s, _Float16* __restrict__ d,
                        int K, int Ncol, int Npad, int n) {
    int i = blockIdx.x * 256 + threadIdx.x;
    if (i >= n) return;
    int k = i / Npad, c = i % Npad;
    d[i] = (k < K && c < Ncol) ? (_Float16)s[(size_t)k * Ncol + c] : (_Float16)0.f;
}

__global__ void k_cvt_act(const float* __restrict__ s, _Float16* __restrict__ d, int n, int relu) {
    int i = blockIdx.x * 256 + threadIdx.x;
    if (i >= n) return;
    float x = s[i];
    if (relu) x = fmaxf(x, 0.f);
    d[i] = (_Float16)x;
}

__global__ void k_copy(const float* __restrict__ s, float* __restrict__ d, int n) {
    int i = blockIdx.x * 256 + threadIdx.x;
    if (i < n) d[i] = s[i];
}

__global__ void k_feats(const float* __restrict__ pos, const float* __restrict__ charge,
                        const float* __restrict__ mask, const float* __restrict__ elem,
                        const float* __restrict__ chars, _Float16* __restrict__ feats) {
    int i = blockIdx.x * 256 + threadIdx.x;
    if (i >= MT_ * FEATP_) return;
    int row = i / FEATP_, c = i % FEATP_;
    float v;
    if (c < 3) v = pos[(size_t)row * 3 + c];
    else if (c == 3) v = charge[row];
    else if (c == 4) v = mask[row];
    else if (c < 133) v = elem[(size_t)row * 128 + (c - 5)];
    else if (c < FEAT_) v = chars[(size_t)row * 256 + (c - 133)];
    else v = 0.f;
    feats[i] = (_Float16)v;
}

// LN(x)*scale+bias -> f16  (wave per row of 128)
__global__ void k_lns(const float* __restrict__ x, const float* __restrict__ sc,
                      const float* __restrict__ bs, _Float16* __restrict__ out) {
    int wid = threadIdx.x >> 5, lane = threadIdx.x & 31;
    int row = blockIdx.x * 8 + wid;
    const float* xr = x + (size_t)row * C_;
    float v[4], s = 0.f;
#pragma unroll
    for (int j = 0; j < 4; ++j) { v[j] = xr[lane + 32 * j]; s += v[j]; }
    for (int o = 16; o > 0; o >>= 1) s += __shfl_xor(s, o);
    float m = s * (1.f / 128.f), vs = 0.f;
#pragma unroll
    for (int j = 0; j < 4; ++j) { float dd = v[j] - m; vs += dd * dd; }
    for (int o = 16; o > 0; o >>= 1) vs += __shfl_xor(vs, o);
    float rstd = rsqrtf(vs * (1.f / 128.f) + 1e-5f);
#pragma unroll
    for (int j = 0; j < 4; ++j) {
        int c = lane + 32 * j;
        out[(size_t)row * C_ + c] = (_Float16)((v[j] - m) * rstd * sc[c] + bs[c]);
    }
}

// an = sigmoid(t1 + b_scale) * LN(a) + t2  -> f16
__global__ void k_adaln_fin(const float* __restrict__ a, const float* __restrict__ t1,
                            const float* __restrict__ bsc, const float* __restrict__ t2,
                            _Float16* __restrict__ anh) {
    int wid = threadIdx.x >> 5, lane = threadIdx.x & 31;
    int row = blockIdx.x * 8 + wid;
    const float* xr = a + (size_t)row * C_;
    float v[4], s = 0.f;
#pragma unroll
    for (int j = 0; j < 4; ++j) { v[j] = xr[lane + 32 * j]; s += v[j]; }
    for (int o = 16; o > 0; o >>= 1) s += __shfl_xor(s, o);
    float m = s * (1.f / 128.f), vs = 0.f;
#pragma unroll
    for (int j = 0; j < 4; ++j) { float dd = v[j] - m; vs += dd * dd; }
    for (int o = 16; o > 0; o >>= 1) vs += __shfl_xor(vs, o);
    float rstd = rsqrtf(vs * (1.f / 128.f) + 1e-5f);
#pragma unroll
    for (int j = 0; j < 4; ++j) {
        int c = lane + 32 * j;
        size_t idx = (size_t)row * C_ + c;
        float g = sigmf(t1[idx] + bsc[c]);
        anh[idx] = (_Float16)(g * (v[j] - m) * rstd + t2[idx]);
    }
}

__global__ void k_residual(float* __restrict__ a, const float* __restrict__ g,
                           const float* __restrict__ bias, const float* __restrict__ o,
                           int n, int ncol) {
    int i = blockIdx.x * 256 + threadIdx.x;
    if (i >= n) return;
    a[i] += sigmf(g[i] + bias[i % ncol]) * o[i];
}

__global__ void k_swiglu(const float* __restrict__ t1, const float* __restrict__ t2,
                         _Float16* __restrict__ out, int n) {
    int i = blockIdx.x * 256 + threadIdx.x;
    if (i >= n) return;
    float x = t1[i];
    out[i] = (_Float16)((x * sigmf(x)) * t2[i]);
}

// gather windowed keys: dst[b,nb,j,c] = src[b, clamp(nb*32-48+j), c] (8 ch / thread)
__global__ void k_gather_h(const _Float16* __restrict__ src, _Float16* __restrict__ dst) {
    int i = blockIdx.x * 256 + threadIdx.x;
    if (i >= B_ * NB_ * NK_ * (C_ / 8)) return;
    int c8 = (i % (C_ / 8)) * 8;
    int j = (i / (C_ / 8)) % NK_;
    int bn = i / ((C_ / 8) * NK_);
    int b = bn / NB_, nb = bn % NB_;
    int ik = nb * NQ_ - 48 + j;
    ik = ik < 0 ? 0 : (ik > N_ - 1 ? N_ - 1 : ik);
    v8h val = *(const v8h*)&src[((size_t)b * N_ + ik) * C_ + c8];
    *(v8h*)&dst[((size_t)bn * NK_ + j) * C_ + c8] = val;
}

// gather + transpose values: dst[b,nb,ch,key] = src[b, clamp(...key), ch]
__global__ void k_gather_vT(const _Float16* __restrict__ src, _Float16* __restrict__ dst) {
    int i = blockIdx.x * 256 + threadIdx.x;
    if (i >= B_ * NB_ * C_ * (NK_ / 8)) return;
    int jt = (i % (NK_ / 8)) * 8;
    int ch = (i / (NK_ / 8)) % C_;
    int bn = i / ((NK_ / 8) * C_);
    int b = bn / NB_, nb = bn % NB_;
    v8h val;
#pragma unroll
    for (int jj = 0; jj < 8; ++jj) {
        int ik = nb * NQ_ - 48 + jt + jj;
        ik = ik < 0 ? 0 : (ik > N_ - 1 ? N_ - 1 : ik);
        val[jj] = src[((size_t)b * N_ + ik) * C_ + ch];
    }
    *(v8h*)&dst[((size_t)bn * C_ + ch) * NK_ + jt] = val;
}

// ======================================================================
// Pair features + 3-layer CP=16 MLP, WMMA f32 16x16x4 per wave tile.
// grid: 16384 blocks of 256 (8 waves); wave = 16 rows of flattened (q,k).
// MLP weights staged transposed in LDS; f32 fragments read as ds_load_b64.
// ======================================================================
__global__ void k_pair(const float* __restrict__ pos, const int* __restrict__ uid,
                       const float* __restrict__ w_d, const float* __restrict__ w_invd,
                       const float* __restrict__ w_vp, const float* __restrict__ clproj,
                       const float* __restrict__ cmproj, const float* __restrict__ w1,
                       const float* __restrict__ w2, const float* __restrict__ w3,
                       float* __restrict__ p_out) {
    __shared__ __align__(16) float baseT[8][16 * 18];
    __shared__ __align__(16) float hT[8][16 * 18];
    __shared__ __align__(16) float wW[3][16 * 18];   // [stage][n*18 + k] (transposed)
    int tid = threadIdx.x, wid = tid >> 5, lane = tid & 31;
    int hi = lane >> 4, li = lane & 15;
    int t = blockIdx.x * 8 + wid;              // tile id, 131072 total
    int chunk = t >> 8;                        // (b,nb)
    int b = chunk / NB_, nb = chunk % NB_;
    int rbase = (t & 255) << 4;                // row base within 4096-row chunk

    // stage the three 16x16 weight matrices (transposed) once per block
    for (int i = tid; i < 3 * 256; i += 256) {
        int s = i >> 8, idx = i & 255, k = idx >> 4, n = idx & 15;
        const float* w = (s == 0) ? w1 : (s == 1) ? w2 : w3;
        wW[s][n * 18 + k] = w[k * CP_ + n];
    }

    int row_in = rbase + li;
    int q = row_in >> 7, kk = row_in & 127;
    int aq = nb * NQ_ + q;
    int ik = nb * NQ_ - 48 + kk;
    int valid = (ik >= 0 && ik < N_);
    int ikc = ik < 0 ? 0 : (ik > N_ - 1 ? N_ - 1 : ik);
    const float* pq = pos + ((size_t)b * N_ + aq) * 3;
    const float* pk = pos + ((size_t)b * N_ + ikc) * 3;
    float d0 = pq[0] - pk[0], d1 = pq[1] - pk[1], d2 = pq[2] - pk[2];
    float inv = 1.f / (1.f + d0 * d0 + d1 * d1 + d2 * d2);
    float vm = (valid && uid[b * N_ + aq] == uid[b * N_ + ikc]) ? 1.f : 0.f;
    const float* clp = clproj + ((size_t)b * N_ + aq) * CP_;
    const float* cmp = cmproj + ((size_t)b * N_ + ikc) * CP_;

    // each lane builds 8 of the 16 pair channels for its row
#pragma unroll
    for (int s = 0; s < 4; ++s)
#pragma unroll
        for (int j = 0; j < 2; ++j) {
            int c = 4 * s + 2 * hi + j;
            float x = (d0 * w_d[c] + d1 * w_d[CP_ + c] + d2 * w_d[2 * CP_ + c] +
                       inv * w_invd[c]) * vm + vm * w_vp[c] + clp[c] + cmp[c];
            baseT[wid][li * 18 + c] = x;
            hT[wid][li * 18 + c] = fmaxf(x, 0.f);
        }
    __syncthreads();

    float accv[8];
    for (int stage = 0; stage < 3; ++stage) {
#if defined(HAVE_WMMA_F32X4)
        v8f acc;
#pragma unroll
        for (int r = 0; r < 8; ++r) acc[r] = 0.f;
#pragma unroll
        for (int s = 0; s < 4; ++s) {
            v2f av = *(const v2f*)&hT[wid][li * 18 + 4 * s + 2 * hi];
            v2f bv = *(const v2f*)&wW[stage][li * 18 + 4 * s + 2 * hi];
            acc = __builtin_amdgcn_wmma_f32_16x16x4_f32(false, av, false, bv,
                                                        (short)0, acc, false, false);
        }
#pragma unroll
        for (int r = 0; r < 8; ++r) accv[r] = acc[r];
#else
#pragma unroll
        for (int r = 0; r < 8; ++r) {
            int rr = r + 8 * hi;
            float s2 = 0.f;
            for (int k2 = 0; k2 < CP_; ++k2)
                s2 += hT[wid][rr * 18 + k2] * wW[stage][li * 18 + k2];
            accv[r] = s2;
        }
#endif
        __syncthreads();
        if (stage < 2) {
#pragma unroll
            for (int r = 0; r < 8; ++r) hT[wid][(r + 8 * hi) * 18 + li] = fmaxf(accv[r], 0.f);
            __syncthreads();
        }
    }
    // residual + store p (C/D layout: row = r + 8*hi, col = li)
#pragma unroll
    for (int r = 0; r < 8; ++r) {
        int rr = r + 8 * hi;
        float out = baseT[wid][rr * 18 + li] + accv[r];
        int row2 = rbase + rr;
        int q2 = row2 >> 7, k2 = row2 & 127;
        p_out[((((size_t)chunk * NQ_ + q2) * NK_) + k2) * CP_ + li] = out;
    }
}

// pair bias: LN_16(p)*sc+bs @ w_pb[16,4]  -> bias[b,nb,h,q,k]
__global__ void k_pairbias(const float* __restrict__ p, const float* __restrict__ sc,
                           const float* __restrict__ bs, const float* __restrict__ wpb,
                           float* __restrict__ biasbf) {
    int i = blockIdx.x * 256 + threadIdx.x;
    if (i >= B_ * NB_ * NQ_ * NK_) return;
    const float* pr = p + (size_t)i * CP_;
    float x[CP_], m = 0.f;
#pragma unroll
    for (int c = 0; c < CP_; ++c) { x[c] = pr[c]; m += x[c]; }
    m *= (1.f / CP_);
    float vs = 0.f;
#pragma unroll
    for (int c = 0; c < CP_; ++c) { float dd = x[c] - m; vs += dd * dd; }
    float rstd = rsqrtf(vs * (1.f / CP_) + 1e-5f);
#pragma unroll
    for (int c = 0; c < CP_; ++c) x[c] = (x[c] - m) * rstd * sc[c] + bs[c];
    int bn = i / (NQ_ * NK_), qk = i % (NQ_ * NK_);
#pragma unroll
    for (int h = 0; h < H_; ++h) {
        float s = 0.f;
#pragma unroll
        for (int c = 0; c < CP_; ++c) s += x[c] * wpb[c * H_ + h];
        biasbf[((size_t)bn * H_ + h) * (NQ_ * NK_) + qk] = s;
    }
}

// logits[b,nb,h,q,k] = (Q Kt)/sqrt(DH); block=(b,nb), wave=head, WMMA K=32
__global__ void k_logits(const _Float16* __restrict__ qh, const _Float16* __restrict__ kgh,
                         float* __restrict__ logits) {
    int bn = blockIdx.x;
    int h = threadIdx.x >> 5, lane = threadIdx.x & 31;
    int hi = lane >> 4, li = lane & 15;
    int b = bn / NB_, nb = bn % NB_;
    size_t qbase = ((size_t)b * N_ + (size_t)nb * NQ_) * C_;
    size_t kbase = (size_t)bn * NK_ * C_;
    int hc = h * DH_;
    V16 aq[2];
#pragma unroll
    for (int mi = 0; mi < 2; ++mi) {
        const _Float16* qp = qh + qbase + (size_t)(mi * 16 + li) * C_ + hc;
        aq[mi].h[0] = *(const v8h*)(qp + hi * 8);
        aq[mi].h[1] = *(const v8h*)(qp + 16 + hi * 8);
    }
    const float scale = 0.1767766952966369f;  // 1/sqrt(32)
    size_t obase = ((size_t)bn * H_ + h) * NQ_ * NK_;
#pragma unroll
    for (int nt = 0; nt < 8; ++nt) {
        int key = nt * 16 + li;
        const _Float16* kp = kgh + kbase + (size_t)key * C_ + hc;
        V16 bk;
        bk.h[0] = *(const v8h*)(kp + hi * 8);
        bk.h[1] = *(const v8h*)(kp + 16 + hi * 8);
#pragma unroll
        for (int mi = 0; mi < 2; ++mi) {
            v8f acc;
#pragma unroll
            for (int r = 0; r < 8; ++r) acc[r] = 0.f;
            acc = __builtin_amdgcn_wmma_f32_16x16x32_f16(false, aq[mi].v, false, bk.v,
                                                         (short)0, acc, false, false);
#pragma unroll
            for (int r = 0; r < 8; ++r) {
                int qr = mi * 16 + r + 8 * hi;
                logits[obase + (size_t)qr * NK_ + key] = acc[r] * scale;
            }
        }
    }
}

// softmax over NK=128, fused +pair-bias +window mask -> attn f16
__global__ void k_softmax(const float* __restrict__ logits, const float* __restrict__ biasbf,
                          _Float16* __restrict__ attnh) {
    int wid = threadIdx.x >> 5, lane = threadIdx.x & 31;
    int rowid = blockIdx.x * 8 + wid;          // over B*NB*H*NQ
    int bn = rowid / (H_ * NQ_);
    int nb = bn % NB_;
    int start = nb * NQ_ - 48;
    float x[4], mx = -3.0e38f;
#pragma unroll
    for (int j = 0; j < 4; ++j) {
        int key = lane + 32 * j;
        int ik = start + key;
        float mb = (ik >= 0 && ik < N_) ? 0.f : -1e9f;
        size_t idx = (size_t)rowid * NK_ + key;
        x[j] = logits[idx] + biasbf[idx] + mb;
        mx = fmaxf(mx, x[j]);
    }
    for (int o = 16; o > 0; o >>= 1) mx = fmaxf(mx, __shfl_xor(mx, o));
    float sum = 0.f;
#pragma unroll
    for (int j = 0; j < 4; ++j) { x[j] = __expf(x[j] - mx); sum += x[j]; }
    for (int o = 16; o > 0; o >>= 1) sum += __shfl_xor(sum, o);
    float r = 1.f / sum;
#pragma unroll
    for (int j = 0; j < 4; ++j)
        attnh[(size_t)rowid * NK_ + lane + 32 * j] = (_Float16)(x[j] * r);
}

// o = attn @ V (V transposed [ch][key]), fused *= sigmoid(g)
__global__ void k_av(const _Float16* __restrict__ attnh, const _Float16* __restrict__ vgT,
                     const float* __restrict__ gf, _Float16* __restrict__ ogh) {
    int bn = blockIdx.x;
    int h = threadIdx.x >> 5, lane = threadIdx.x & 31;
    int hi = lane >> 4, li = lane & 15;
    int b = bn / NB_, nb = bn % NB_;
    size_t abase = ((size_t)bn * H_ + h) * NQ_ * NK_;
    size_t vbase = (size_t)bn * C_ * NK_;
    v8f acc[2][2];
#pragma unroll
    for (int mi = 0; mi < 2; ++mi)
#pragma unroll
        for (int ni = 0; ni < 2; ++ni)
#pragma unroll
            for (int r = 0; r < 8; ++r) acc[mi][ni][r] = 0.f;

    for (int ks = 0; ks < 4; ++ks) {
        V16 af[2];
#pragma unroll
        for (int mi = 0; mi < 2; ++mi) {
            const _Float16* ap = attnh + abase + (size_t)(mi * 16 + li) * NK_ + ks * 32;
            af[mi].h[0] = *(const v8h*)(ap + hi * 8);
            af[mi].h[1] = *(const v8h*)(ap + 16 + hi * 8);
        }
#pragma unroll
        for (int ni = 0; ni < 2; ++ni) {
            int ch = h * DH_ + ni * 16 + li;
            const _Float16* vp = vgT + vbase + (size_t)ch * NK_ + ks * 32;
            V16 bf;
            bf.h[0] = *(const v8h*)(vp + hi * 8);
            bf.h[1] = *(const v8h*)(vp + 16 + hi * 8);
#pragma unroll
            for (int mi = 0; mi < 2; ++mi)
                acc[mi][ni] = __builtin_amdgcn_wmma_f32_16x16x32_f16(false, af[mi].v, false,
                                                                     bf.v, (short)0,
                                                                     acc[mi][ni], false, false);
        }
    }
    size_t rowbase = (size_t)b * N_ + (size_t)nb * NQ_;
#pragma unroll
    for (int mi = 0; mi < 2; ++mi)
#pragma unroll
        for (int ni = 0; ni < 2; ++ni)
#pragma unroll
            for (int r = 0; r < 8; ++r) {
                int m = mi * 16 + r + 8 * hi;
                int ch = h * DH_ + ni * 16 + li;
                size_t o = (rowbase + m) * C_ + ch;
                ogh[o] = (_Float16)(acc[mi][ni][r] * sigmf(gf[o]));
            }
}

// segment mean over sorted atom_to_token_idx: block = (b,t), binary search range
__device__ __forceinline__ int lbound(const int* a, int n, int v) {
    int lo = 0, hi = n;
    while (lo < hi) { int mid = (lo + hi) >> 1; if (a[mid] < v) lo = mid + 1; else hi = mid; }
    return lo;
}

__global__ void k_segmean(const float* __restrict__ tok, const int* __restrict__ a2t,
                          float* __restrict__ out) {
    int b = blockIdx.x / T_, t = blockIdx.x % T_;
    const int* arr = a2t + (size_t)b * N_;
    int lo = lbound(arr, N_, t), hi = lbound(arr, N_, t + 1);
    int cnt = hi - lo;
    float inv = 1.f / (float)(cnt > 0 ? cnt : 1);
    for (int c = threadIdx.x; c < CT_; c += 128) {
        float s = 0.f;
        for (int r = lo; r < hi; ++r) s += tok[((size_t)b * N_ + r) * CT_ + c];
        out[((size_t)b * T_ + t) * CT_ + c] = s * inv;
    }
}

// ======================================================================
// host orchestration
// ======================================================================
// params pytree (dicts sorted by key): per-block 26 leaves, tail 10 leaves
#define IN_P0 7

extern "C" void kernel_launch(void* const* d_in, const int* in_sizes, int n_in,
                              void* d_out, int out_size, void* d_ws, size_t ws_size,
                              hipStream_t stream) {
    (void)in_sizes; (void)n_in; (void)out_size; (void)ws_size;
    const float* ref_pos    = (const float*)d_in[0];
    const float* ref_charge = (const float*)d_in[1];
    const float* ref_mask   = (const float*)d_in[2];
    const float* ref_elem   = (const float*)d_in[3];
    const float* ref_chars  = (const float*)d_in[4];
    const int*   uid        = (const int*)d_in[5];
    const int*   a2t        = (const int*)d_in[6];
    auto P = [&](int i) { return (const float*)d_in[i]; };

    char* ws = (char*)d_ws;
    size_t off = 0;
    auto alloc = [&](size_t bytes) -> char* {
        off = (off + 255) & ~(size_t)255;
        char* p = ws + off;
        off += bytes;
        return p;
    };

    // d_out regions (a_token, a, c_l, p)
    float* o_atok = (float*)d_out;
    float* o_a    = o_atok + (size_t)B_ * T_ * CT_;
    float* o_cl   = o_a + (size_t)MT_ * C_;
    float* o_p    = o_cl + (size_t)MT_ * C_;

    // ---- f16 weight conversion (rows padded to Kpad, cols padded to x64) ----
    auto cvtW = [&](const float* src, int K, int Kpad, int Ncol) -> _Float16* {
        int Npad = (Ncol + 63) & ~63;
        _Float16* dst = (_Float16*)alloc((size_t)Kpad * Npad * 2);
        int n = Kpad * Npad;
        k_cvt_w<<<(n + 255) / 256, 256, 0, stream>>>(src, dst, K, Ncol, Npad, n);
        return dst;
    };
    _Float16* wf_h = cvtW(P(IN_P0 + 84), FEAT_, FEATP_, C_);
    struct BW { _Float16 *wsA, *wbA, *wsT, *wbT, *wq, *wk, *wv, *wg, *wo, *wog, *wtg, *wt1, *wt2, *wt3; } bw[NBLK_];
    for (int b = 0; b < NBLK_; ++b) {
        int base = IN_P0 + 26 * b;
        bw[b].wsA = cvtW(P(base + 4), C_, C_, C_);   // ada_attn.w_scale
        bw[b].wbA = cvtW(P(base + 3), C_, C_, C_);   // ada_attn.w_bias
        bw[b].wsT = cvtW(P(base + 9), C_, C_, C_);   // ada_tr.w_scale
        bw[b].wbT = cvtW(P(base + 8), C_, C_, C_);   // ada_tr.w_bias
        bw[b].wq  = cvtW(P(base + 20), C_, C_, C_);
        bw[b].wk  = cvtW(P(base + 16), C_, C_, C_);
        bw[b].wv  = cvtW(P(base + 25), C_, C_, C_);
        bw[b].wg  = cvtW(P(base + 15), C_, C_, C_);
        bw[b].wo  = cvtW(P(base + 17), C_, C_, C_);
        bw[b].wog = cvtW(P(base + 18), C_, C_, C_);
        bw[b].wtg = cvtW(P(base + 24), C_, C_, C_);
        bw[b].wt1 = cvtW(P(base + 21), C_, C_, 2 * C_);
        bw[b].wt2 = cvtW(P(base + 22), C_, C_, 2 * C_);
        bw[b].wt3 = cvtW(P(base + 23), 2 * C_, 2 * C_, C_);
    }
    _Float16* wcl_h = cvtW(P(IN_P0 + 81), C_, C_, CP_);
    _Float16* wcm_h = cvtW(P(IN_P0 + 82), C_, C_, CP_);
    _Float16* wqo_h = cvtW(P(IN_P0 + 86), C_, C_, CT_);

    // ---- scratch ----
    _Float16* feats_h = (_Float16*)alloc((size_t)MT_ * FEATP_ * 2);
    _Float16* cl_h  = (_Float16*)alloc((size_t)MT_ * C_ * 2);
    _Float16* cr_h  = (_Float16*)alloc((size_t)MT_ * C_ * 2);
    _Float16* sn_h  = (_Float16*)alloc((size_t)MT_ * C_ * 2);
    _Float16* an_h  = (_Float16*)alloc((size_t)MT_ * C_ * 2);
    _Float16* q_h   = (_Float16*)alloc((size_t)MT_ * C_ * 2);
    _Float16* k_h   = (_Float16*)alloc((size_t)MT_ * C_ * 2);
    _Float16* v_h   = (_Float16*)alloc((size_t)MT_ * C_ * 2);
    _Float16* og_h  = (_Float16*)alloc((size_t)MT_ * C_ * 2);
    _Float16* a_h   = (_Float16*)alloc((size_t)MT_ * C_ * 2);
    _Float16* hbig_h = (_Float16*)alloc((size_t)MT_ * 2 * C_ * 2);
    _Float16* kg_h  = (_Float16*)alloc((size_t)B_ * NB_ * NK_ * C_ * 2);
    _Float16* vgT_h = (_Float16*)alloc((size_t)B_ * NB_ * C_ * NK_ * 2);
    _Float16* attn_h = (_Float16*)alloc((size_t)B_ * NB_ * H_ * NQ_ * NK_ * 2);
    float* t1f    = (float*)alloc((size_t)MT_ * C_ * 4);
    float* t2f    = (float*)alloc((size_t)MT_ * C_ * 4);
    float* gf     = (float*)alloc((size_t)MT_ * C_ * 4);
    float* clproj = (float*)alloc((size_t)MT_ * CP_ * 4);
    float* cmproj = (float*)alloc((size_t)MT_ * CP_ * 4);
    float* logits = (float*)alloc((size_t)B_ * NB_ * H_ * NQ_ * NK_ * 4);  // 33.5MB
    float* biasbf = (float*)alloc((size_t)B_ * NB_ * H_ * NQ_ * NK_ * 4);  // 33.5MB
    float* t1big = logits;   // disjoint lifetime reuse (16.8MB each)
    float* t2big = biasbf;
    float* tok   = logits;   // 25.2MB, used after attention

    auto gemm = [&](const _Float16* A, const _Float16* W, const float* bias,
                    float* outF, _Float16* outH, int M, int K, int Nc, int relu) {
        int ldw = (Nc + 63) & ~63;
        dim3 g((M + 63) / 64, (Nc + 63) / 64);
        k_gemm<<<g, 128, 0, stream>>>(A, W, bias, outF, outH, M, K, Nc, ldw, relu);
    };

    // ---- embed + pair features ----
    k_feats<<<(MT_ * FEATP_ + 255) / 256, 256, 0, stream>>>(ref_pos, ref_charge, ref_mask,
                                                            ref_elem, ref_chars, feats_h);
    gemm(feats_h, wf_h, nullptr, o_cl, cl_h, MT_, FEATP_, C_, 0);
    k_cvt_act<<<(MT_ * C_ + 255) / 256, 256, 0, stream>>>(o_cl, cr_h, MT_ * C_, 1);
    k_copy<<<(MT_ * C_ + 255) / 256, 256, 0, stream>>>(o_cl, o_a, MT_ * C_);  // a = c_l
    gemm(cr_h, wcl_h, nullptr, clproj, nullptr, MT_, C_, CP_, 0);
    gemm(cr_h, wcm_h, nullptr, cmproj, nullptr, MT_, C_, CP_, 0);
    k_pair<<<16384, 256, 0, stream>>>(ref_pos, uid, P(IN_P0 + 83), P(IN_P0 + 85),
                                      P(IN_P0 + 87), clproj, cmproj, P(IN_P0 + 78),
                                      P(IN_P0 + 79), P(IN_P0 + 80), o_p);

    // ---- transformer blocks ----
    for (int b = 0; b < NBLK_; ++b) {
        int base = IN_P0 + 26 * b;
        // AdaLN (attention)
        k_lns<<<MT_ / 8, 256, 0, stream>>>(o_cl, P(base + 2), P(base + 1), sn_h);
        gemm(sn_h, bw[b].wsA, nullptr, t1f, nullptr, MT_, C_, C_, 0);
        gemm(sn_h, bw[b].wbA, nullptr, t2f, nullptr, MT_, C_, C_, 0);
        k_adaln_fin<<<MT_ / 8, 256, 0, stream>>>(o_a, t1f, P(base + 0), t2f, an_h);
        // QKV + gate
        gemm(an_h, bw[b].wq, P(base + 11), nullptr, q_h, MT_, C_, C_, 0);
        gemm(an_h, bw[b].wk, nullptr, nullptr, k_h, MT_, C_, C_, 0);
        gemm(an_h, bw[b].wv, nullptr, nullptr, v_h, MT_, C_, C_, 0);
        gemm(an_h, bw[b].wg, nullptr, gf, nullptr, MT_, C_, C_, 0);
        k_gather_h<<<(B_ * NB_ * NK_ * (C_ / 8) + 255) / 256, 256, 0, stream>>>(k_h, kg_h);
        k_gather_vT<<<(B_ * NB_ * C_ * (NK_ / 8) + 255) / 256, 256, 0, stream>>>(v_h, vgT_h);
        // pair bias + attention
        k_pairbias<<<(B_ * NB_ * NQ_ * NK_ + 255) / 256, 256, 0, stream>>>(
            o_p, P(base + 14), P(base + 13), P(base + 19), biasbf);
        k_logits<<<B_ * NB_, 128, 0, stream>>>(q_h, kg_h, logits);
        k_softmax<<<B_ * NB_ * H_ * NQ_ / 8, 256, 0, stream>>>(logits, biasbf, attn_h);
        k_av<<<B_ * NB_, 128, 0, stream>>>(attn_h, vgT_h, gf, og_h);
        gemm(og_h, bw[b].wo, nullptr, t1f, nullptr, MT_, C_, C_, 0);
        gemm(cl_h, bw[b].wog, nullptr, t2f, nullptr, MT_, C_, C_, 0);
        k_residual<<<(MT_ * C_ + 255) / 256, 256, 0, stream>>>(o_a, t2f, P(base + 10),
                                                               t1f, MT_ * C_, C_);
        // conditioned transition (SwiGLU)
        k_lns<<<MT_ / 8, 256, 0, stream>>>(o_cl, P(base + 7), P(base + 6), sn_h);
        gemm(sn_h, bw[b].wsT, nullptr, t1f, nullptr, MT_, C_, C_, 0);
        gemm(sn_h, bw[b].wbT, nullptr, t2f, nullptr, MT_, C_, C_, 0);
        k_adaln_fin<<<MT_ / 8, 256, 0, stream>>>(o_a, t1f, P(base + 5), t2f, an_h);
        gemm(an_h, bw[b].wt1, nullptr, t1big, nullptr, MT_, C_, 2 * C_, 0);
        gemm(an_h, bw[b].wt2, nullptr, t2big, nullptr, MT_, C_, 2 * C_, 0);
        k_swiglu<<<(MT_ * 2 * C_ + 255) / 256, 256, 0, stream>>>(t1big, t2big, hbig_h,
                                                                 MT_ * 2 * C_);
        gemm(hbig_h, bw[b].wt3, nullptr, t1f, nullptr, MT_, 2 * C_, C_, 0);
        gemm(cl_h, bw[b].wtg, nullptr, t2f, nullptr, MT_, C_, C_, 0);
        k_residual<<<(MT_ * C_ + 255) / 256, 256, 0, stream>>>(o_a, t2f, P(base + 12),
                                                               t1f, MT_ * C_, C_);
    }

    // ---- token aggregation ----
    k_cvt_act<<<(MT_ * C_ + 255) / 256, 256, 0, stream>>>(o_a, a_h, MT_ * C_, 0);
    gemm(a_h, wqo_h, nullptr, tok, nullptr, MT_, C_, CT_, 1 /*relu*/);
    k_segmean<<<B_ * T_, 128, 0, stream>>>(tok, a2t, o_atok);
}
